// EMB_Loss_90726889160818
// MI455X (gfx1250) — matrix operand order repaired and checked by reading, not compiled, not run on hardware
//
#include <hip/hip_runtime.h>
#include <math.h>

// Problem constants (from reference)
#define NB 8
#define NC 32
#define ND 256
#define WH 25600                 // 160*160
#define QTILE 256                // pixels per workgroup tile
#define DCHUNK 64                // d rows staged in LDS per chunk
#define NCHUNKS (ND / DCHUNK)    // 4
#define LDS_PAD 8                // bf16 elements of row padding (16B) -> conflict-free strides
#define MSTRIDE (QTILE + LDS_PAD)
#define USTRIDE (QTILE + LDS_PAD)

typedef __bf16 bf16;
typedef __attribute__((ext_vector_type(16))) __bf16 v16bf;
typedef __attribute__((ext_vector_type(8)))  float  v8f;

// ---------------------------------------------------------------------------
// Zero the workspace accumulators (s[32*256] + n[32])
// ---------------------------------------------------------------------------
__global__ void emb_zero(float* __restrict__ ws) {
    int i = blockIdx.x * blockDim.x + threadIdx.x;
    if (i < NC * ND + NC) ws[i] = 0.0f;
}

// ---------------------------------------------------------------------------
// Main kernel: per-pixel normalize + s[c,d] = sum_p mask[c,p]*U[d,p] via WMMA
// grid = NB * (WH/QTILE) = 800 blocks, 256 threads (8 wave32)
// ---------------------------------------------------------------------------
__global__ __launch_bounds__(256) void emb_main(const float* __restrict__ feat,
                                                const float* __restrict__ mask,
                                                float* __restrict__ s_acc,
                                                float* __restrict__ n_acc) {
    __shared__ bf16 mask_lds[NC * MSTRIDE];     // 16.9 KB, class-major bf16 tile
    __shared__ bf16 u_lds[DCHUNK * USTRIDE];    // 33.8 KB, d-major normalized tile

    const int t    = threadIdx.x;
    const int blk  = blockIdx.x;
    const int b    = blk / (WH / QTILE);
    const int q0   = (blk % (WH / QTILE)) * QTILE;

    const long long fbase = (long long)b * ND * WH + q0 + t;   // feat[(b*ND+d)*WH + q]
    const long long mbase = (long long)b * NC * WH + q0 + t;   // mask[(b*NC+c)*WH + q]

    // ---- Phase A: per-pixel inverse norm (thread t owns pixel q0+t) --------
    float ss = 0.0f;
#pragma unroll 4
    for (int d = 0; d < ND; ++d) {
        float v = feat[fbase + (long long)d * WH];
        ss = fmaf(v, v, ss);
    }
    const float invn = 1.0f / fmaxf(sqrtf(ss), 1e-8f);

    // ---- Phase B: stage mask tile as bf16 (exact for {0,1}) ----------------
#pragma unroll 4
    for (int c = 0; c < NC; ++c) {
        float mv = mask[mbase + (long long)c * WH];
        mask_lds[c * MSTRIDE + t] = (bf16)mv;
    }
    __syncthreads();

    // per-class pixel count for this tile (32 threads, one class each)
    if (t < NC) {
        float cnt = 0.0f;
        for (int q = 0; q < QTILE; ++q) cnt += (float)mask_lds[t * MSTRIDE + q];
        atomicAdd(&n_acc[t], cnt);
    }

    // ---- Phase C: WMMA contraction over pixels -----------------------------
    const int wave = t >> 5;
    const int lane = t & 31;
    const int hi   = lane >> 4;     // 0: lanes 0-15, 1: lanes 16-31
    const int lrow = lane & 15;
    const int ct   = wave & 1;      // class tile (2 tiles of 16)
    const int dt   = wave >> 1;     // d tile within chunk (4 tiles of 16)

    const bf16* __restrict__ arow = &mask_lds[(ct * 16 + lrow) * MSTRIDE];
    const bf16* __restrict__ brow = &u_lds[(dt * 16 + lrow) * USTRIDE];

    for (int ch = 0; ch < NCHUNKS; ++ch) {
        // restage U chunk: u_lds[dl][q] = feat[ch*64+dl][q] * invn[q]
        // (second read of feat; L2-hot since phase A just streamed this tile)
#pragma unroll 4
        for (int dl = 0; dl < DCHUNK; ++dl) {
            float v = feat[fbase + (long long)(ch * DCHUNK + dl) * WH];
            u_lds[dl * USTRIDE + t] = (bf16)(v * invn);
        }
        __syncthreads();

        // A = mask tile 16(c) x 32(q), B = U tile 32(q) x 16(d), C/D f32 16x16
        v8f acc = {};
#pragma unroll
        for (int k = 0; k < QTILE; k += 32) {
            v16bf afrag, bfrag;
#pragma unroll
            for (int sl = 0; sl < 8; ++sl) {
                // 16-bit A layout: slots 0-3 -> K 0..7(+8 for hi lanes),
                //                  slots 4-7 -> K 16..23(+8 for hi lanes)
                int ka = k + ((sl < 4) ? (2 * sl) : (16 + 2 * (sl - 4))) + hi * 8;
                afrag[2 * sl]     = arow[ka];
                afrag[2 * sl + 1] = arow[ka + 1];
                // 16-bit B layout: same K mapping as A (table order)
                int kb = k + ((sl < 4) ? (2 * sl) : (16 + 2 * (sl - 4))) + hi * 8;
                bfrag[2 * sl]     = brow[kb];
                bfrag[2 * sl + 1] = brow[kb + 1];
            }
            acc = __builtin_amdgcn_wmma_f32_16x16x32_bf16(
                false, afrag, false, bfrag, (short)0, acc, false, false);
        }

        // C/D layout: lane half selects M+8, vgpr index = M low 3 bits, N = lane&15
#pragma unroll
        for (int i = 0; i < 8; ++i) {
            int cg = ct * 16 + i + hi * 8;
            int dg = ch * DCHUNK + dt * 16 + lrow;
            atomicAdd(&s_acc[cg * ND + dg], acc[i]);
        }
        __syncthreads();   // before next chunk overwrites u_lds
    }
}

// ---------------------------------------------------------------------------
// Epilogue: one wave32, lane c handles class c, shuffle-reduce, exp(-2*mean)
// ---------------------------------------------------------------------------
__global__ __launch_bounds__(32) void emb_final(const float* __restrict__ s_acc,
                                                const float* __restrict__ n_acc,
                                                float* __restrict__ out) {
    int c = threadIdx.x;            // 0..31
    float n = n_acc[c];
    float ss = 0.0f;
    for (int d = 0; d < ND; ++d) {
        float v = s_acc[c * ND + d];
        ss = fmaf(v, v, ss);
    }
    float pair  = 0.5f * (ss - n);
    float denom = 0.5f * n * (n + 1.0f);
    float cl = (n > 0.0f) ? (pair / fmaxf(denom, 1.0f)) : 0.0f;
    for (int off = 16; off > 0; off >>= 1)
        cl += __shfl_down(cl, off, 32);
    if (c == 0)
        out[0] = expf(-2.0f * (cl / (float)NC));   // ALPHA=1, GAMMA=2
}

// ---------------------------------------------------------------------------
extern "C" void kernel_launch(void* const* d_in, const int* in_sizes, int n_in,
                              void* d_out, int out_size, void* d_ws, size_t ws_size,
                              hipStream_t stream) {
    const float* feat = (const float*)d_in[0];   // [8,256,160,160] f32
    const float* mask = (const float*)d_in[1];   // [8,32,160,160] f32
    float* s_acc = (float*)d_ws;                 // [32*256]
    float* n_acc = s_acc + NC * ND;              // [32]
    float* out   = (float*)d_out;                // scalar

    emb_zero<<<(NC * ND + NC + 255) / 256, 256, 0, stream>>>(s_acc);
    emb_main<<<NB * (WH / QTILE), 256, 0, stream>>>(feat, mask, s_acc, n_acc);
    emb_final<<<1, 32, 0, stream>>>(s_acc, n_acc, out);
}
